// RWKV8FeedForward_45337674777327
// MI455X (gfx1250) — matrix operand order repaired and checked by reading, not compiled, not run on hardware
//
#include <hip/hip_runtime.h>
#include <stdint.h>

// Problem dims (fixed by the reference)
#define DIMC 2048          // model dim (power of two: >>11)
#define HID  1792          // hidden dim = 14*128 = 56*32
#define BB   4
#define TT   4096          // power of two: >>12
#define MM   (BB*TT)       // 16384 rows
#define SEG  64
#define NSEG (TT/SEG)      // 64
#define LN_EPS 1e-5f

// LDS tile geometry: 128 rows x 32 K-elements, row padded to 40 bf16 (80 B)
// -> bank = (20*row)%64, 16 distinct banks across a fragment's 16 lanes.
#define AROW 40
#define TILE_HALF_B  (64 * AROW * 2)    // +64 rows, bytes
#define TILE_BUF_B   (128 * AROW * 2)   // one buffer, bytes (10240)

typedef __attribute__((ext_vector_type(16))) __bf16 v16bf;
typedef __attribute__((ext_vector_type(8)))  __bf16 v8bf;
typedef __attribute__((ext_vector_type(8)))  float  v8f;

union ABFrag { v16bf v; v8bf h[2]; };

// ---------------------------------------------------------------------------
// CDNA5 async global->LDS copy (ASYNCcnt path), 16 B per lane.
// GV mode: 64-bit global address in VGPR pair, LDS byte offset in VGPR.
// ---------------------------------------------------------------------------
__device__ __forceinline__ void async_b128(uint32_t lds_off, const __bf16* g) {
  asm volatile("global_load_async_to_lds_b128 %0, %1, off"
               :: "v"(lds_off), "v"(g)
               : "memory");
}
__device__ __forceinline__ void wait_async_le4() {
  asm volatile("s_wait_asynccnt 0x4" ::: "memory");
}
__device__ __forceinline__ void wait_async_0() {
  asm volatile("s_wait_asynccnt 0x0" ::: "memory");
}

// ---------------------------------------------------------------------------
// WMMA fragment loaders from an LDS tile (bf16 16x16x32, ISA 7.12.2 layouts).
// A (16x32): lane<16 -> K [0..7]+[16..23] of row m+lane; lane>=16 -> +8.
// B (32x16): column n = LDS row n; lane<16 -> K [0..7]+[16..23]; lane>=16 -> +8.
// All reads are 16B-aligned v8bf (ds_load_b128), conflict-free via AROW pad.
// ---------------------------------------------------------------------------
__device__ __forceinline__ v16bf lds_a_frag(const __bf16* tile, int m, int lane) {
  const int row = m + (lane & 15);
  const int kb  = (lane >> 4) << 3;       // 0 or 8
  const __bf16* p = tile + row * AROW + kb;
  ABFrag f;
  f.h[0] = *(const v8bf*)(p);             // K kb .. kb+7
  f.h[1] = *(const v8bf*)(p + 16);        // K kb+16 .. kb+23
  return f.v;
}

__device__ __forceinline__ v16bf lds_b_frag(const __bf16* tile, int n, int lane) {
  const int col = n + (lane & 15);
  const int kb  = (lane >> 4) << 3;       // 0 or 8
  const __bf16* p = tile + col * AROW + kb;
  ABFrag f;
  f.h[0] = *(const v8bf*)(p);             // K kb .. kb+7
  f.h[1] = *(const v8bf*)(p + 16);        // K kb+16 .. kb+23
  return f.v;
}

// ---------------------------------------------------------------------------
// GEMM: C[M,N] (f32) = A[M,K] (bf16, row-major) * W[N,K]^T (bf16, row-major)
// Block = 256 threads = 8 waves (2 x 4); wave tile = 64(M) x 32(N);
// block tile = 128 x 128; K stepped by 32 with v_wmma_f32_16x16x32_bf16.
// Tiles staged through LDS with double-buffered async copies.
// ---------------------------------------------------------------------------
__global__ __launch_bounds__(256) void gemm_bf16_nt(
    const __bf16* __restrict__ A, const __bf16* __restrict__ W,
    float* __restrict__ C, int M, int N, int K) {
  __shared__ __bf16 As[2][128 * AROW];
  __shared__ __bf16 Bs[2][128 * AROW];

  const int tid  = threadIdx.x;
  const int lane = tid & 31;
  const int wave = tid >> 5;
  const int block_m = blockIdx.y * 128;
  const int block_n = blockIdx.x * 128;
  const int wm = (wave >> 2) * 64;   // wave's M offset inside block tile
  const int wn = (wave & 3) * 32;    // wave's N offset inside block tile

  // Cooperative fill mapping: thread t copies 16B chunks of rows r0 and r0+64
  // (for both A and B tiles). 256 threads x 4 chunks = 2 x 8KB per stage.
  const int r0 = tid >> 2;               // 0..63
  const int c8 = (tid & 3) * 8;          // K sub-offset 0,8,16,24

  const __bf16* gA0 = A + (size_t)(block_m + r0) * K + c8;
  const __bf16* gA1 = A + (size_t)(block_m + r0 + 64) * K + c8;
  const __bf16* gB0 = W + (size_t)(block_n + r0) * K + c8;
  const __bf16* gB1 = W + (size_t)(block_n + r0 + 64) * K + c8;

  const uint32_t chunk = (uint32_t)((r0 * AROW + c8) * 2);
  const uint32_t aoff  = (uint32_t)(uintptr_t)(&As[0][0]) + chunk;
  const uint32_t boff  = (uint32_t)(uintptr_t)(&Bs[0][0]) + chunk;

  const int S = K >> 5;   // number of 32-wide K stages

  // Prefetch stage 0 into buffer 0
  async_b128(aoff, gA0);
  async_b128(aoff + TILE_HALF_B, gA1);
  async_b128(boff, gB0);
  async_b128(boff + TILE_HALF_B, gB1);

  v8f acc[4][2] = {};

  for (int s = 0; s < S; ++s) {
    const int buf = s & 1;
    if (s + 1 < S) {
      const int kn = (s + 1) << 5;
      const uint32_t bo = (uint32_t)((s + 1) & 1) * TILE_BUF_B;
      async_b128(aoff + bo, gA0 + kn);
      async_b128(aoff + bo + TILE_HALF_B, gA1 + kn);
      async_b128(boff + bo, gB0 + kn);
      async_b128(boff + bo + TILE_HALF_B, gB1 + kn);
      wait_async_le4();   // in-order: remaining <=4 are the newest -> stage s done
    } else {
      wait_async_0();
    }
    __syncthreads();      // all waves' stage-s tile data visible in LDS

    const __bf16* At = &As[buf][0];
    const __bf16* Bt = &Bs[buf][0];
    v16bf a[4], b[2];
#pragma unroll
    for (int i = 0; i < 4; ++i) a[i] = lds_a_frag(At, wm + i * 16, lane);
#pragma unroll
    for (int j = 0; j < 2; ++j) b[j] = lds_b_frag(Bt, wn + j * 16, lane);
#pragma unroll
    for (int i = 0; i < 4; ++i)
#pragma unroll
      for (int j = 0; j < 2; ++j)
        acc[i][j] = __builtin_amdgcn_wmma_f32_16x16x32_bf16(
            false, a[i], false, b[j], (short)0, acc[i][j], false, false);

    __syncthreads();      // protect buffer before next iteration's async fill
  }

  // C/D layout: VGPR r -> M = r (lanes 0-15) or 8+r (lanes 16-31), N = lane%16
  const int col_off = lane & 15;
  const int row_off = (lane >> 4) << 3;
#pragma unroll
  for (int i = 0; i < 4; ++i)
#pragma unroll
    for (int j = 0; j < 2; ++j) {
      const int col = block_n + wn + j * 16 + col_off;
      size_t idx = (size_t)(block_m + wm + i * 16 + row_off) * N + col;
#pragma unroll
      for (int r = 0; r < 8; ++r) {
        C[idx] = acc[i][j][r];
        idx += (size_t)N;
      }
    }
}

// ---------------------------------------------------------------------------
// Token shift + time-mix, fused with f32->bf16 conversion of the 3 A operands
// ---------------------------------------------------------------------------
__global__ void mix_shift_kernel(const float* __restrict__ x,
                                 const float* __restrict__ tmk,
                                 const float* __restrict__ tmr,
                                 const float* __restrict__ tmw,
                                 __bf16* __restrict__ kin,
                                 __bf16* __restrict__ rin,
                                 __bf16* __restrict__ win) {
  const size_t idx = (size_t)blockIdx.x * blockDim.x + threadIdx.x;
  if (idx >= (size_t)MM * DIMC) return;
  const int    c = (int)(idx & (DIMC - 1));
  const size_t m = idx >> 11;             // / DIMC
  const int    t = (int)(m & (TT - 1));
  const size_t b = m >> 12;               // / TT
  const int   ts = (t == 0) ? 1 : t - 1;  // reference uses x[:,1:2] for row 0
  const float xv = x[idx];
  const float xs = x[(((size_t)b * TT + (size_t)ts) << 11) + c];
  const float dx = xs - xv;
  kin[idx] = (__bf16)(xv + dx * tmk[c]);
  rin[idx] = (__bf16)(xv + dx * tmr[c]);
  win[idx] = (__bf16)(xv + dx * tmw[c]);
}

__global__ void f32_to_bf16_kernel(const float* __restrict__ src,
                                   __bf16* __restrict__ dst, size_t n) {
  const size_t i = (size_t)blockIdx.x * blockDim.x + threadIdx.x;
  if (i < n) dst[i] = (__bf16)src[i];
}

// ---------------------------------------------------------------------------
// Segmented parallel scan of state = state*decay + relu(k)^2
// Pass 1: per-segment local scan with zero init -> seg[b,s,h]
// ---------------------------------------------------------------------------
__global__ __launch_bounds__(256) void scan_pass1_kernel(
    const float* __restrict__ k, float* __restrict__ seg,
    const float* __restrict__ time_decay) {
  const int h = blockIdx.x * blockDim.x + threadIdx.x;
  if (h >= HID) return;
  const int s = blockIdx.y;
  const int b = blockIdx.z;
  const float decay = __expf(-__expf(time_decay[h]));
  float st = 0.f;
  size_t i = ((size_t)b * TT + (size_t)s * SEG) * HID + h;
  for (int t = 0; t < SEG; ++t) {
    float kv = fmaxf(k[i], 0.f);
    st = st * decay + kv * kv;
    i += HID;
  }
  seg[((size_t)b * NSEG + s) * HID + h] = st;
}

// Pass 2: serial prefix over the (64) segments per channel; rewrite seg[b,s,h]
// with the *incoming* state for segment s.  decay^SEG by 6 squarings.
__global__ __launch_bounds__(256) void scan_pass2_kernel(
    float* __restrict__ seg, const float* __restrict__ time_decay) {
  const int h = blockIdx.x * blockDim.x + threadIdx.x;
  if (h >= HID) return;
  const int b = blockIdx.y;
  const float decay = __expf(-__expf(time_decay[h]));
  float dL = decay;
#pragma unroll
  for (int i = 0; i < 6; ++i) dL *= dL;   // decay^64
  float carry = 0.f;
  const size_t base = (size_t)b * NSEG * HID + h;
  for (int s = 0; s < NSEG; ++s) {
    const size_t idx = base + (size_t)s * HID;
    const float local = seg[idx];
    seg[idx] = carry;
    carry = carry * dL + local;
  }
}

// Pass 3: re-scan each segment with correct init, fused with
// out = state * r * sigmoid(w), converted to bf16 (A operand of final GEMM).
__global__ __launch_bounds__(256) void scan_pass3_kernel(
    const float* __restrict__ k, const float* __restrict__ r,
    const float* __restrict__ w, const float* __restrict__ seg,
    const float* __restrict__ time_decay, __bf16* __restrict__ outa) {
  const int h = blockIdx.x * blockDim.x + threadIdx.x;
  if (h >= HID) return;
  const int s = blockIdx.y;
  const int b = blockIdx.z;
  const float decay = __expf(-__expf(time_decay[h]));
  float st = seg[((size_t)b * NSEG + s) * HID + h];
  size_t i = ((size_t)b * TT + (size_t)s * SEG) * HID + h;
  for (int t = 0; t < SEG; ++t) {
    float kv = fmaxf(k[i], 0.f);
    st = st * decay + kv * kv;
    const float gate = 1.f / (1.f + __expf(-w[i]));
    outa[i] = (__bf16)(st * r[i] * gate);
    i += HID;
  }
}

// ---------------------------------------------------------------------------
// In-place layernorm over the last dim (one block per row)
// ---------------------------------------------------------------------------
__global__ __launch_bounds__(256) void layernorm_kernel(
    float* __restrict__ y, const float* __restrict__ lw,
    const float* __restrict__ lb) {
  __shared__ float sh[16];
  __shared__ float stats[2];
  float* p = y + (size_t)blockIdx.x * DIMC;
  float s = 0.f, s2 = 0.f;
  for (int i = threadIdx.x; i < DIMC; i += 256) {
    const float v = p[i];
    s += v; s2 += v * v;
  }
#pragma unroll
  for (int off = 16; off; off >>= 1) {
    s  += __shfl_down(s,  off, 32);
    s2 += __shfl_down(s2, off, 32);
  }
  const int lane = threadIdx.x & 31, wave = threadIdx.x >> 5;
  if (lane == 0) { sh[wave] = s; sh[8 + wave] = s2; }
  __syncthreads();
  if (threadIdx.x == 0) {
    float ts = 0.f, ts2 = 0.f;
#pragma unroll
    for (int i = 0; i < 8; ++i) { ts += sh[i]; ts2 += sh[8 + i]; }
    const float mu  = ts / (float)DIMC;
    const float var = ts2 / (float)DIMC - mu * mu;
    stats[0] = mu;
    stats[1] = rsqrtf(var + LN_EPS);
  }
  __syncthreads();
  const float mu = stats[0], inv = stats[1];
  for (int i = threadIdx.x; i < DIMC; i += 256)
    p[i] = (p[i] - mu) * inv * lw[i] + lb[i];
}

// ---------------------------------------------------------------------------
extern "C" void kernel_launch(void* const* d_in, const int* in_sizes, int n_in,
                              void* d_out, int out_size, void* d_ws, size_t ws_size,
                              hipStream_t stream) {
  const float* x   = (const float*)d_in[0];
  const float* tmk = (const float*)d_in[1];
  const float* tmr = (const float*)d_in[2];
  const float* tmw = (const float*)d_in[3];
  const float* Wk  = (const float*)d_in[4];
  const float* Wr  = (const float*)d_in[5];
  const float* Wg  = (const float*)d_in[6];
  const float* Wv  = (const float*)d_in[7];
  const float* td  = (const float*)d_in[8];
  const float* lnw = (const float*)d_in[9];
  const float* lnb = (const float*)d_in[10];
  float* out = (float*)d_out;

  // Workspace carve-up (256B aligned slices)
  char* ws = (char*)d_ws;
  size_t off = 0;
  auto carve = [&](size_t bytes) -> char* {
    char* p = ws + off;
    off += (bytes + 255) & ~(size_t)255;
    return p;
  };
  __bf16* kin  = (__bf16*)carve((size_t)MM * DIMC * 2);
  __bf16* rin  = (__bf16*)carve((size_t)MM * DIMC * 2);
  __bf16* win  = (__bf16*)carve((size_t)MM * DIMC * 2);
  __bf16* Wk_b = (__bf16*)carve((size_t)HID * DIMC * 2);
  __bf16* Wr_b = (__bf16*)carve((size_t)HID * DIMC * 2);
  __bf16* Wg_b = (__bf16*)carve((size_t)HID * DIMC * 2);
  __bf16* Wv_b = (__bf16*)carve((size_t)DIMC * HID * 2);
  float*  kb   = (float*)carve((size_t)MM * HID * 4);
  float*  rb   = (float*)carve((size_t)MM * HID * 4);
  float*  wb   = (float*)carve((size_t)MM * HID * 4);
  __bf16* outa = (__bf16*)carve((size_t)MM * HID * 2);
  float*  segb = (float*)carve((size_t)BB * NSEG * HID * 4);
  (void)ws_size; (void)in_sizes; (void)n_in; (void)out_size;

  // 1) Weight conversion to bf16
  {
    const size_t nw = (size_t)HID * DIMC;
    const int g = (int)((nw + 255) / 256);
    f32_to_bf16_kernel<<<g, 256, 0, stream>>>(Wk, Wk_b, nw);
    f32_to_bf16_kernel<<<g, 256, 0, stream>>>(Wr, Wr_b, nw);
    f32_to_bf16_kernel<<<g, 256, 0, stream>>>(Wg, Wg_b, nw);
    f32_to_bf16_kernel<<<g, 256, 0, stream>>>(Wv, Wv_b, nw);
  }

  // 2) Token shift + mix -> bf16 A operands
  {
    const size_t n = (size_t)MM * DIMC;
    mix_shift_kernel<<<(int)(n / 256), 256, 0, stream>>>(
        x, tmk, tmr, tmw, kin, rin, win);
  }

  // 3) Three WMMA GEMMs: [16384,2048] x [2048,1792]
  {
    dim3 grid(HID / 128, MM / 128);
    gemm_bf16_nt<<<grid, 256, 0, stream>>>(kin, Wk_b, kb, MM, HID, DIMC);
    gemm_bf16_nt<<<grid, 256, 0, stream>>>(rin, Wr_b, rb, MM, HID, DIMC);
    gemm_bf16_nt<<<grid, 256, 0, stream>>>(win, Wg_b, wb, MM, HID, DIMC);
  }

  // 4) Segmented scan (3 passes), fused elementwise -> bf16 A operand
  {
    dim3 g1(HID / 256, NSEG, BB);
    scan_pass1_kernel<<<g1, 256, 0, stream>>>(kb, segb, td);
    dim3 g2(HID / 256, BB);
    scan_pass2_kernel<<<g2, 256, 0, stream>>>(segb, td);
    scan_pass3_kernel<<<g1, 256, 0, stream>>>(kb, rb, wb, segb, td, outa);
  }

  // 5) Final WMMA GEMM: [16384,1792] x [1792,2048] -> f32 d_out
  {
    dim3 grid(DIMC / 128, MM / 128);
    gemm_bf16_nt<<<grid, 256, 0, stream>>>(outa, Wv_b, out, MM, DIMC, HID);
  }

  // 6) In-place layernorm over last dim
  layernorm_kernel<<<MM, 256, 0, stream>>>(out, lnw, lnb);
}
